// MAGF_37641093382741
// MI455X (gfx1250) — compile-verified, matrix operand order checked by reading, not run on hardware
//
#include <hip/hip_runtime.h>
#include <math.h>

#define WAYN 5
#define NQN  150
#define CIN  640
#define CE   64
#define NPIX 100
#define NIMG 155

typedef __attribute__((ext_vector_type(16))) __bf16 v16bf;
typedef __attribute__((ext_vector_type(8)))  float  v8f;
typedef __attribute__((ext_vector_type(4)))  unsigned int u32x4;

union FragU {
  v16bf bf;
  u32x4 q[2];
  unsigned short us[16];
};

// ---- workspace byte offsets ----
#define WS_W1BF   0            // 64*640 bf16            (81920 B)
#define WS_RSUM   81920        // 64 f32                 (256 B)
#define WS_A2     82176        // 5*16*32 bf16           (5120 B)
#define WS_CONST  87296        // 240 f32                (960 B)
#define WS_MEANS  88320        // 155*100 f32            (62000 B)
#define WS_EMB    150336       // 155*100*64 bf16        (1984000 B)

// ---- k_main LDS byte offsets ----
#define MO_X      0            // 10000 f32
#define MO_A1     40000        // 10000 f32 (aliases semb/qemb in time)
#define MO_SEMB   40000        // 112*64 bf16
#define MO_QEMB   54336        // 112*64 bf16
#define MO_B1     80000        // 10000 f32
#define MO_X2C    120000       // 10*168*16 bf16 (53760 B)
#define MO_YP     173760       // 3*16000 bf16   (96000 B)
#define MO_OUT    269760       // 10000 f32
#define MO_CST    309760       // 240 f32
#define MO_ATTNS  310720       // 100 f32
#define MO_ATTNQ  311120       // 100 f32
#define MO_RED    311520       // 4 f32
#define SMEM_MAIN 311552

#define SMEM_EMB  (112*CIN*2 + NPIX*CE*4 + 512)

__device__ __forceinline__ unsigned short f2bf(float x) {
  unsigned u = __builtin_bit_cast(unsigned, x);
  unsigned r = (u + 0x7FFFu + ((u >> 16) & 1u)) >> 16;
  return (unsigned short)r;
}
__device__ __forceinline__ float bf2f(unsigned short h) {
  unsigned u = ((unsigned)h) << 16;
  return __builtin_bit_cast(float, u);
}

// Load a 16x32 bf16 A/B fragment. Storage is row-major [row][stride] bf16
// ("row" = M for A, = N for B since B is stored [n][k]).
// ISA 16-bit layout: lanes 0-15 hold K {k0..k0+7, k0+16..k0+23},
// lanes 16-31 hold K {k0+8..k0+15, k0+24..k0+31} (two 16B chunks per lane).
__device__ __forceinline__ v16bf load_frag16(const unsigned short* base, int row0,
                                             int k0, int stride) {
  int lane = threadIdx.x & 31;
  int m = lane & 15, half = lane >> 4;
  const unsigned short* r = base + (row0 + m) * stride + k0 + half * 8;
  FragU f;
  f.q[0] = *(const u32x4*)(r);
  f.q[1] = *(const u32x4*)(r + 16);
  return f.bf;
}

// ============================ k_prep ===================================
__global__ void __launch_bounds__(256) k_prep(
    const float* w1x1, const float* s1c2, const float* s1c1,
    const float* s1bna_s, const float* s1bna_b,
    const float* s1bnb_s, const float* s1bnb_b,
    const float* s1proj, const float* s1bnp_s, const float* s1bnp_b,
    const float* s2c2, const float* s2bna_s, const float* s2bna_b,
    const float* s2c1, const float* s2bnb_s, const float* s2bnb_b,
    const float* s2proj, const float* s2bnp_s, const float* s2bnp_b,
    char* ws) {
  int tid = threadIdx.x;
  unsigned short* w1bf = (unsigned short*)(ws + WS_W1BF);
  float* rsum = (float*)(ws + WS_RSUM);
  unsigned short* a2 = (unsigned short*)(ws + WS_A2);
  float* cst = (float*)(ws + WS_CONST);

  for (int i = tid; i < CE * CIN; i += 256) w1bf[i] = f2bf(w1x1[i]);
  if (tid < CE) {
    float s = 0.f;
    for (int c = 0; c < CIN; ++c) s += w1x1[tid * CIN + c];
    rsum[tid] = s;
  }
  // Stage-2 conv_uv weights as 5 tap-pair matrices [16o][32k] bf16:
  // k<16 -> tap 2p channel k ; k>=16 -> tap 2p+1 channel k-16 (tap 9 = zero).
  for (int i = tid; i < 5 * 16 * 32; i += 256) {
    int p = i >> 9, rem = i & 511, o = rem >> 5, k = rem & 31;
    int tap = (k < 16) ? 2 * p : 2 * p + 1;
    int ic = k & 15;
    float v = 0.f;
    if (tap < 9) v = s2c2[((o * 16 + ic) * 3 + tap / 3) * 3 + tap % 3];
    a2[i] = f2bf(v);
  }
  // Folded stage-2 conv_hw+proj weights: weff[i][dh][dw]
  for (int i = tid; i < 144; i += 256) {
    int ic = i / 9, t = i % 9;
    float s = 0.f;
    for (int o = 0; o < 16; ++o)
      s += s2proj[o] * s2bnb_s[o] * s2c1[(o * 16 + ic) * 9 + t];
    cst[96 + i] = s;
  }
  if (tid < 16) {
    cst[32 + tid] = s1bnp_s[tid] * s1proj[tid];  // a_i
    cst[48 + tid] = s1bnp_b[tid];                // c_i
    cst[64 + tid] = s2bna_s[tid];
    cst[80 + tid] = s2bna_b[tid];
  }
  if (tid == 0) {
    for (int i = 0; i < 9; ++i) { cst[i] = s1c2[i]; cst[9 + i] = s1c1[i]; }
    cst[18] = s1bna_s[0]; cst[19] = s1bna_b[0];
    cst[20] = s1bnb_s[0]; cst[21] = s1bnb_b[0];
    float cb = 0.f;
    for (int o = 0; o < 16; ++o) cb += s2proj[o] * s2bnb_b[o];
    cst[22] = cb;
    cst[23] = s2bnp_s[0]; cst[24] = s2bnp_b[0];
    for (int i = 25; i < 32; ++i) cst[i] = 0.f;
  }
}

// ============================ k_embed ==================================
// One block per image: y = relu(bn0(W(x - mean))) then L2-normalize over CE.
// GEMM [64,640]x[640,100] via bf16 WMMA; mean folded via rowsum(W).
__global__ void __launch_bounds__(256) k_embed(const float* spt, const float* qry,
                                               const float* bn0s, const float* bn0b,
                                               char* ws) {
  extern __shared__ char smem[];
  unsigned short* xbf = (unsigned short*)smem;                 // [112][640] bf16
  float* ybuf = (float*)(smem + 112 * CIN * 2);                // [100][64] f32
  float* meanL = (float*)(smem + 112 * CIN * 2 + NPIX * CE * 4);  // [100]

  int tid = threadIdx.x;
  int img = blockIdx.x;
  const float* src = (img < WAYN) ? (spt + (size_t)img * CIN * NPIX)
                                  : (qry + (size_t)(img - WAYN) * CIN * NPIX);
  const unsigned short* w1bf = (const unsigned short*)(ws + WS_W1BF);
  const float* rsum = (const float*)(ws + WS_RSUM);
  float* meansG = (float*)(ws + WS_MEANS) + (size_t)img * NPIX;
  unsigned short* embG = (unsigned short*)(ws + WS_EMB) + (size_t)img * NPIX * CE;

  if (tid < NPIX) {
    float s = 0.f;
    for (int c = 0; c < CIN; ++c) s += src[c * NPIX + tid];
    float m = s * (1.0f / CIN);
    meanL[tid] = m;
    meansG[tid] = m;
  }
  for (int i = tid; i < 112 * CIN; i += 256) {
    int p = i / CIN, c = i - p * CIN;
    float v = (p < NPIX) ? src[c * NPIX + p] : 0.f;
    xbf[i] = f2bf(v);
  }
  __syncthreads();

  int wv = tid >> 5, lane = tid & 31, lm = lane & 15, half = lane >> 4;
  for (int t = wv; t < 28; t += 8) {           // 4 M-tiles x 7 N-tiles
    int mt = t / 7, nt = t % 7;
    v8f acc = {0.f, 0.f, 0.f, 0.f, 0.f, 0.f, 0.f, 0.f};
    for (int kc = 0; kc < 20; ++kc) {
      v16bf a = load_frag16(w1bf, mt * 16, kc * 32, CIN);   // A from global (L2)
      v16bf b = load_frag16(xbf, nt * 16, kc * 32, CIN);    // B from LDS
      acc = __builtin_amdgcn_wmma_f32_16x16x32_bf16(false, a, false, b,
                                                    (short)0, acc, false, false);
    }
    int p = nt * 16 + lm;
#pragma unroll
    for (int r = 0; r < 8; ++r) {
      int o = mt * 16 + r + 8 * half;
      if (p < NPIX) {
        float y = acc[r] - rsum[o] * meanL[p];
        y = fmaxf(bn0s[o] * y + bn0b[o], 0.f);
        ybuf[p * CE + o] = y;
      }
    }
  }
  __syncthreads();
  if (tid < NPIX) {
    float s = 0.f;
    for (int o = 0; o < CE; ++o) { float y = ybuf[tid * CE + o]; s += y * y; }
    float inv = 1.0f / fmaxf(sqrtf(s), 1e-8f);
    for (int o = 0; o < CE; ++o)
      embG[tid * CE + o] = f2bf(ybuf[tid * CE + o] * inv);   // [pixel][channel]
  }
}

// ============================ k_main ===================================
// One block per (q, way): corr GEMM -> stage1 convs -> stage2 conv_uv (WMMA)
// -> folded conv_hw+proj -> gnorm/softmax attention -> cosine similarity.
__global__ void __launch_bounds__(256) k_main(const float* spt, const float* qry,
                                              char* ws, float* out) {
  extern __shared__ char smem[];
  float* xbuf = (float*)(smem + MO_X);
  float* A1 = (float*)(smem + MO_A1);
  unsigned short* semb = (unsigned short*)(smem + MO_SEMB);
  unsigned short* qemb = (unsigned short*)(smem + MO_QEMB);
  float* B1 = (float*)(smem + MO_B1);
  unsigned short* X2C = (unsigned short*)(smem + MO_X2C);
  unsigned short* Yp = (unsigned short*)(smem + MO_YP);
  float* obuf = (float*)(smem + MO_OUT);
  float* cst = (float*)(smem + MO_CST);
  float* attS = (float*)(smem + MO_ATTNS);
  float* attQ = (float*)(smem + MO_ATTNQ);
  float* red = (float*)(smem + MO_RED);

  int tid = threadIdx.x;
  int b = blockIdx.x;
  int w = b % WAYN, q = b / WAYN;
  int wv = tid >> 5, lane = tid & 31, lm = lane & 15, half = lane >> 4;

  const float* wsc = (const float*)(ws + WS_CONST);
  if (tid < 240) cst[tid] = wsc[tid];

  // Preload the 5 tap-pair A matrices into register fragments (global).
  const unsigned short* a2g = (const unsigned short*)(ws + WS_A2);
  v16bf a2f[5];
#pragma unroll
  for (int p = 0; p < 5; ++p) {
    const unsigned short* r = a2g + p * 512 + lm * 32 + half * 8;
    FragU f;
    f.q[0] = *(const u32x4*)(r);
    f.q[1] = *(const u32x4*)(r + 16);
    a2f[p] = f.bf;
  }

  // Load both embeddings into LDS (zero-pad rows 100..111).
  const unsigned short* embG = (const unsigned short*)(ws + WS_EMB);
  const unsigned short* eS = embG + (size_t)w * NPIX * CE;
  const unsigned short* eQ = embG + (size_t)(WAYN + q) * NPIX * CE;
  for (int i = tid; i < 112 * CE; i += 256) {
    int row = i >> 6;
    unsigned short vs = (row < NPIX) ? eS[i] : (unsigned short)0;
    unsigned short vq = (row < NPIX) ? eQ[i] : (unsigned short)0;
    semb[i] = vs;
    qemb[i] = vq;
  }
  __syncthreads();

  // ---- corr[uv][hw] = sum_c s[uv][c]*q[hw][c], K=64 (2 WMMA per tile) ----
  for (int t = wv; t < 49; t += 8) {
    int mt = t / 7, nt = t % 7;
    v8f acc = {0.f, 0.f, 0.f, 0.f, 0.f, 0.f, 0.f, 0.f};
#pragma unroll
    for (int kc = 0; kc < 2; ++kc) {
      v16bf a = load_frag16(semb, mt * 16, kc * 32, CE);
      v16bf bb = load_frag16(qemb, nt * 16, kc * 32, CE);
      acc = __builtin_amdgcn_wmma_f32_16x16x32_bf16(false, a, false, bb,
                                                    (short)0, acc, false, false);
    }
    int hw = nt * 16 + lm;
#pragma unroll
    for (int r = 0; r < 8; ++r) {
      int uv = mt * 16 + r + 8 * half;
      if (uv < NPIX && hw < NPIX) xbuf[uv * NPIX + hw] = acc[r];
    }
  }
  __syncthreads();

  // ---- stage1: 3x3 conv over (u,v), BN, ReLU ----
  for (int i = tid; i < 10000; i += 256) {
    int uv = i / 100, hw = i - uv * 100;
    int u = uv / 10, v = uv - u * 10;
    float acc = 0.f;
    for (int du = -1; du <= 1; ++du) {
      int uu = u + du; if (uu < 0 || uu > 9) continue;
      for (int dv = -1; dv <= 1; ++dv) {
        int vv = v + dv; if (vv < 0 || vv > 9) continue;
        acc += cst[(du + 1) * 3 + dv + 1] * xbuf[(uu * 10 + vv) * 100 + hw];
      }
    }
    A1[i] = fmaxf(cst[18] * acc + cst[19], 0.f);
  }
  __syncthreads();
  // ---- stage1: 3x3 conv over (h,w), BN (no relu) -> B1 ----
  for (int i = tid; i < 10000; i += 256) {
    int uv = i / 100, hw = i - uv * 100;
    int h = hw / 10, wc = hw - h * 10;
    float acc = 0.f;
    for (int dh = -1; dh <= 1; ++dh) {
      int hh = h + dh; if (hh < 0 || hh > 9) continue;
      for (int dw = -1; dw <= 1; ++dw) {
        int ww = wc + dw; if (ww < 0 || ww > 9) continue;
        acc += cst[9 + (dh + 1) * 3 + dw + 1] * A1[uv * 100 + hh * 10 + ww];
      }
    }
    B1[i] = cst[20] * acc + cst[21];
  }
  // zero Yp slot 2 (represents plane h = -1)
  for (int i = tid; i < 16000; i += 256) Yp[2 * 16000 + i] = 0;
  __syncthreads();

  // ---- stage2: rolling h-plane pipeline ----
  for (int hp = 0; hp <= 10; ++hp) {
    if (hp < 10) {
      // Build halo-padded X2 columns for hw = hp*10 + w, w=0..9:
      // X2C[w][(u'+1)*12+(v'+1)][i] = relu(a_i*B1[uv', hw] + c_i), zero outside.
      for (int i = tid; i < 10 * 168 * 16; i += 256) {
        int ww = i / 2688, rem = i - ww * 2688;
        int row = rem >> 4, ic = rem & 15;
        int iu = row / 12, iv = row - iu * 12;
        int u = iu - 1, v = iv - 1;
        float val = 0.f;
        if (u >= 0 && u < 10 && v >= 0 && v < 10)
          val = fmaxf(cst[32 + ic] * B1[(u * 10 + v) * 100 + hp * 10 + ww] +
                          cst[48 + ic], 0.f);
        X2C[i] = f2bf(val);
      }
      __syncthreads();
      // WMMA conv over (u,v): Y1[o][uv] += W2[o, i@tapA | i@tapB] * X2 cols.
      unsigned short* Ys = Yp + (hp % 3) * 16000;
      for (int t = wv; t < 70; t += 8) {          // 10 w-cols x 7 uv-tiles
        int ww = t / 7, uvt = t % 7, uv0 = uvt * 16;
        const unsigned short* Xw = X2C + ww * 2688;
        int n = uv0 + lm;
        int u = n / 10, v = n - u * 10;
        v8f acc = {0.f, 0.f, 0.f, 0.f, 0.f, 0.f, 0.f, 0.f};
#pragma unroll
        for (int p = 0; p < 5; ++p) {
          int ta = 2 * p, tb = 2 * p + 1;
          int rowA = (u + ta / 3) * 12 + (v + ta % 3);
          int rowB = (tb < 9) ? (u + tb / 3) * 12 + (v + tb % 3) : 167; // zero row
          FragU f;
          f.q[0] = *(const u32x4*)(Xw + rowA * 16 + half * 8);
          f.q[1] = *(const u32x4*)(Xw + rowB * 16 + half * 8);
          acc = __builtin_amdgcn_wmma_f32_16x16x32_bf16(false, a2f[p], false, f.bf,
                                                        (short)0, acc, false, false);
        }
        int uv = uv0 + lm;
#pragma unroll
        for (int r = 0; r < 8; ++r) {
          int o = r + 8 * half;
          if (uv < NPIX) {
            float y = fmaxf(cst[64 + o] * acc[r] + cst[80 + o], 0.f);
            Ys[(o * 100 + uv) * 10 + ww] = f2bf(y);
          }
        }
      }
    } else {
      for (int i = tid; i < 16000; i += 256) Yp[(10 % 3) * 16000 + i] = 0; // h=10
    }
    __syncthreads();
    if (hp >= 1) {
      int r = hp - 1;  // consume output row h = r
      for (int e = tid; e < 1000; e += 256) {
        int uv = e / 10, ww = e - uv * 10;
        float acc = cst[22];  // folded bnb2 bias
        for (int ic = 0; ic < 16; ++ic) {
          for (int dh = -1; dh <= 1; ++dh) {
            int hh = r + dh;
            int slot = ((hh % 3) + 3) % 3;
            const unsigned short* Ysl = Yp + slot * 16000;
            const float* wrow = &cst[96 + (ic * 3 + dh + 1) * 3];
            for (int dw = -1; dw <= 1; ++dw) {
              int wn = ww + dw; if (wn < 0 || wn > 9) continue;
              acc += wrow[dw + 1] * bf2f(Ysl[(ic * 100 + uv) * 10 + wn]);
            }
          }
        }
        obuf[uv * 100 + r * 10 + ww] = cst[23] * acc + cst[24];
      }
    }
    __syncthreads();
  }

  // ---- gnorm + softmax attention ----
  if (tid < 100) attS[tid] = 0.f;
  else if (tid < 200) attQ[tid - 100] = 0.f;
  else if (tid < 204) red[tid - 200] = 0.f;
  __syncthreads();
  if (tid < 100) {  // softmax over uv for column hw=tid
    int col = tid;
    float s = 0.f, ss = 0.f;
    for (int i = 0; i < 100; ++i) { float x = obuf[i * 100 + col]; s += x; ss += x * x; }
    float m = s * 0.01f;
    float var = (ss - 100.f * m * m) * (1.f / 99.f);
    float inv = 1.f / (sqrtf(var + 1e-5f) * 5.0f);  // /T_ATTN
    float zmax = -1e30f;
    for (int i = 0; i < 100; ++i)
      zmax = fmaxf(zmax, (obuf[i * 100 + col] - m) * inv);
    float se = 0.f;
    for (int i = 0; i < 100; ++i)
      se += __expf((obuf[i * 100 + col] - m) * inv - zmax);
    float rinv = 1.f / se;
    for (int i = 0; i < 100; ++i) {
      float pz = __expf((obuf[i * 100 + col] - m) * inv - zmax) * rinv;
      atomicAdd(&attS[i], pz);
    }
  } else if (tid >= 128 && tid < 228) {  // softmax over hw for row uv
    int row = tid - 128;
    float s = 0.f, ss = 0.f;
    for (int j = 0; j < 100; ++j) { float x = obuf[row * 100 + j]; s += x; ss += x * x; }
    float m = s * 0.01f;
    float var = (ss - 100.f * m * m) * (1.f / 99.f);
    float inv = 1.f / (sqrtf(var + 1e-5f) * 5.0f);
    float zmax = -1e30f;
    for (int j = 0; j < 100; ++j)
      zmax = fmaxf(zmax, (obuf[row * 100 + j] - m) * inv);
    float se = 0.f;
    for (int j = 0; j < 100; ++j)
      se += __expf((obuf[row * 100 + j] - m) * inv - zmax);
    float rinv = 1.f / se;
    for (int j = 0; j < 100; ++j) {
      float pz = __expf((obuf[row * 100 + j] - m) * inv - zmax) * rinv;
      atomicAdd(&attQ[j], pz);
    }
  }
  __syncthreads();

  // ---- attention pooling + cosine similarity ----
  const float* meansG = (const float*)(ws + WS_MEANS);
  const float* msp = meansG + w * NPIX;
  const float* mqp = meansG + (WAYN + q) * NPIX;
  float dotp = 0.f, na2 = 0.f, nb2 = 0.f;
  for (int c = tid; c < CIN; c += 256) {
    const float* srow = spt + ((size_t)w * CIN + c) * NPIX;
    const float* qrow = qry + ((size_t)q * CIN + c) * NPIX;
    float sp = 0.f, qp = 0.f;
    for (int p = 0; p < 100; ++p) {
      sp += attS[p] * (srow[p] - msp[p]);
      qp += attQ[p] * (qrow[p] - mqp[p]);
    }
    sp *= 0.01f; qp *= 0.01f;
    dotp += sp * qp; na2 += sp * sp; nb2 += qp * qp;
  }
  atomicAdd(&red[0], dotp);
  atomicAdd(&red[1], na2);
  atomicAdd(&red[2], nb2);
  __syncthreads();
  if (tid == 0) {
    float na = fmaxf(sqrtf(red[1]), 1e-8f);
    float nb = fmaxf(sqrtf(red[2]), 1e-8f);
    out[q * WAYN + w] = red[0] / (na * nb) * 5.0f;  // /TEMP, TEMP=0.2
  }
}

// ============================ launch ===================================
extern "C" void kernel_launch(void* const* d_in, const int* in_sizes, int n_in,
                              void* d_out, int out_size, void* d_ws, size_t ws_size,
                              hipStream_t stream) {
  const float* spt = (const float*)d_in[0];
  const float* qry = (const float*)d_in[1];
  const float* w1x1 = (const float*)d_in[2];
  const float* bn0s = (const float*)d_in[3];
  const float* bn0b = (const float*)d_in[4];
  const float* s1c2 = (const float*)d_in[5];
  const float* s1bna_s = (const float*)d_in[6];
  const float* s1bna_b = (const float*)d_in[7];
  const float* s1c1 = (const float*)d_in[8];
  const float* s1bnb_s = (const float*)d_in[9];
  const float* s1bnb_b = (const float*)d_in[10];
  const float* s1proj = (const float*)d_in[11];
  const float* s1bnp_s = (const float*)d_in[12];
  const float* s1bnp_b = (const float*)d_in[13];
  const float* s2c2 = (const float*)d_in[14];
  const float* s2bna_s = (const float*)d_in[15];
  const float* s2bna_b = (const float*)d_in[16];
  const float* s2c1 = (const float*)d_in[17];
  const float* s2bnb_s = (const float*)d_in[18];
  const float* s2bnb_b = (const float*)d_in[19];
  const float* s2proj = (const float*)d_in[20];
  const float* s2bnp_s = (const float*)d_in[21];
  const float* s2bnp_b = (const float*)d_in[22];
  char* ws = (char*)d_ws;

  (void)hipFuncSetAttribute((const void*)k_embed,
                            hipFuncAttributeMaxDynamicSharedMemorySize, SMEM_EMB);
  (void)hipFuncSetAttribute((const void*)k_main,
                            hipFuncAttributeMaxDynamicSharedMemorySize, SMEM_MAIN);

  k_prep<<<1, 256, 0, stream>>>(w1x1, s1c2, s1c1, s1bna_s, s1bna_b, s1bnb_s,
                                s1bnb_b, s1proj, s1bnp_s, s1bnp_b, s2c2, s2bna_s,
                                s2bna_b, s2c1, s2bnb_s, s2bnb_b, s2proj, s2bnp_s,
                                s2bnp_b, ws);
  k_embed<<<NIMG, 256, SMEM_EMB, stream>>>(spt, qry, bn0s, bn0b, ws);
  k_main<<<NQN * WAYN, 256, SMEM_MAIN, stream>>>(spt, qry, ws, (float*)d_out);
}
